// LSTMClassifier_70351564309199
// MI455X (gfx1250) — compile-verified, matrix operand order checked by reading, not run on hardware
//
#include <hip/hip_runtime.h>
#include <hip/hip_bf16.h>
#include <math.h>

// ---------------- problem dims ----------------
#define B_   64
#define T_   256
#define EMB_ 512
#define RNN_ 1024
#define HID_ 128
#define LAB_ 4
#define NG_  (4 * RNN_)   // 4096 gate columns

typedef __attribute__((ext_vector_type(16))) __bf16 bf16x16;
typedef __attribute__((ext_vector_type(8)))  __bf16 bf16x8;
typedef __attribute__((ext_vector_type(8)))  float  f32x8;

union FragAB { bf16x16 v; bf16x8 h[2]; };

__device__ __forceinline__ __bf16 f2bf(float f) {
    union { float f; unsigned u; } v; v.f = f;
    unsigned r = v.u + 0x7FFFu + ((v.u >> 16) & 1u);   // round-to-nearest-even
    unsigned short hs = (unsigned short)(r >> 16);
    return __builtin_bit_cast(__bf16, hs);
}
__device__ __forceinline__ float bf2f(__bf16 b) {
    unsigned u = ((unsigned)__builtin_bit_cast(unsigned short, b)) << 16;
    return __builtin_bit_cast(float, u);
}
__device__ __forceinline__ float sigm(float x) { return 1.0f / (1.0f + __expf(-x)); }

// ---------------- 1) embedding gather + cast, layout (t, b, k) ----------------
__global__ void gather_cast_kernel(const int* __restrict__ x,
                                   const float* __restrict__ emb,
                                   __bf16* __restrict__ eT) {
    size_t idx = (size_t)blockIdx.x * blockDim.x + threadIdx.x;   // over B*T*EMB
    int k   = (int)(idx & (EMB_ - 1));
    int row = (int)(idx >> 9);          // t*B + b
    int t = row >> 6;                   // B_ = 64
    int b = row & (B_ - 1);
    int tok = x[b * T_ + t];
    eT[idx] = f2bf(emb[(size_t)tok * EMB_ + k]);
}

// ---------------- 2) transpose + cast weights: out[n][k] = in[k][n] ----------------
__global__ void transpose_cast_kernel(const float* __restrict__ in,
                                      __bf16* __restrict__ out,
                                      int K, int N) {   // in is K x N, out is N x K
    size_t idx = (size_t)blockIdx.x * blockDim.x + threadIdx.x;   // over N*K (n-major)
    int k = (int)(idx % K);
    int n = (int)(idx / K);
    out[(size_t)n * K + k] = f2bf(in[(size_t)k * N + n]);
}

// ---------------- 3) persistent LSTM recurrence, WMMA bf16 ----------------
// grid = B/16 workgroups, block = 512 threads (16 waves).
// Wave w owns RNN slice j in [w*64, w*64+64), processed as two 32-wide halves.
// For each half, its 4 gate columns r*1024 + j land in the wave's own accumulators.
__global__ __launch_bounds__(512)
void lstm_kernel(const int* __restrict__ x,
                 const __bf16* __restrict__ eT,     // (T, B, EMB) bf16
                 const __bf16* __restrict__ Wt,     // (NG, EMB)  bf16 (transposed)
                 const __bf16* __restrict__ Ut,     // (NG, RNN)  bf16 (transposed)
                 const float*  __restrict__ bias,   // (NG,) f32
                 float* __restrict__ pooled) {      // (B, RNN) f32
    __shared__ __bf16 hsh[2][16 * RNN_];            // 2 x 32 KB ping-pong h (A source)
    __shared__ float  xm[2][16];                    // mask  (x!=0), double buffered
    __shared__ float  xp[2][16];                    // pool  (x==2), double buffered

    const int tid   = threadIdx.x;
    const int lane  = tid & 31;
    const int w     = tid >> 5;                     // wave id 0..15
    const int b0    = blockIdx.x * 16;
    const int lcol  = lane & 15;                    // N / M-column within tile
    const int kc    = (lane >> 4) << 3;             // 0 or 8 : K sub-chunk select
    const int mbase = (lane >> 4) << 3;             // accum rows: M = v + mbase

    // zero h buffer 0 (read at t=0)
    for (int i = tid; i < 16 * RNN_; i += 512)
        hsh[0][i] = __builtin_bit_cast(__bf16, (unsigned short)0);

    // per-(half, sub) j columns and row-base pointers into transposed weights
    const __bf16* wls[2][2];
    const __bf16* uls[2][2];
    float bia[2][4][2];
#pragma unroll
    for (int hf = 0; hf < 2; ++hf)
#pragma unroll
        for (int s = 0; s < 2; ++s) {
            const int j = w * 64 + hf * 32 + s * 16 + lcol;
            wls[hf][s] = Wt + (size_t)j * EMB_ + kc;   // gate r adds r*RNN_*EMB_ (imm offset)
            uls[hf][s] = Ut + (size_t)j * RNN_ + kc;   // gate r adds r*RNN_*RNN_ (imm offset)
#pragma unroll
            for (int r = 0; r < 4; ++r)
                bia[hf][r][s] = bias[r * RNN_ + j];
        }

    float c_[2][2][8] = {};    // cell state   [half][sub][row]
    float p_[2][2][8] = {};    // pooled accum [half][sub][row]

    for (int t = 0; t < T_; ++t) {
        const int cur = t & 1, nxt = cur ^ 1;
        if (tid < 16) {
            int xv = x[(b0 + tid) * T_ + t];
            xm[cur][tid] = (xv != 0) ? 1.0f : 0.0f;
            xp[cur][tid] = (xv == 2) ? 1.0f : 0.0f;
        }
        __syncthreads();   // prev-step h writes + this-step masks visible

        float mk[8], pk[8];
#pragma unroll
        for (int v = 0; v < 8; ++v) { mk[v] = xm[cur][v + mbase]; pk[v] = xp[cur][v + mbase]; }

        const __bf16* abase = eT + ((size_t)t * B_ + b0 + lcol) * EMB_ + kc;
        const __bf16* hbase = &hsh[cur][lcol * RNN_ + kc];

#pragma unroll
        for (int hf = 0; hf < 2; ++hf) {
            f32x8 acc[4][2] = {};

            // ---- phase 1: z += e_t @ W   (K = 512) ----
#pragma unroll 1
            for (int k0 = 0; k0 < EMB_; k0 += 32) {
                FragAB a, bb[8];
                a.h[0] = *(const bf16x8*)(abase + k0);
                a.h[1] = *(const bf16x8*)(abase + k0 + 16);
#pragma unroll
                for (int r = 0; r < 4; ++r)
#pragma unroll
                    for (int s = 0; s < 2; ++s) {
                        const __bf16* p = wls[hf][s] + (size_t)r * (RNN_ * EMB_) + k0;
                        bb[r * 2 + s].h[0] = *(const bf16x8*)(p);
                        bb[r * 2 + s].h[1] = *(const bf16x8*)(p + 16);
                    }
#pragma unroll
                for (int r = 0; r < 4; ++r)
#pragma unroll
                    for (int s = 0; s < 2; ++s)
                        acc[r][s] = __builtin_amdgcn_wmma_f32_16x16x32_bf16(
                            false, a.v, false, bb[r * 2 + s].v, (short)0, acc[r][s], false, false);
            }

            // ---- phase 2: z += h @ U    (K = 1024), h from LDS bf16 ----
#pragma unroll 1
            for (int k0 = 0; k0 < RNN_; k0 += 32) {
                FragAB a, bb[8];
                a.h[0] = *(const bf16x8*)(hbase + k0);
                a.h[1] = *(const bf16x8*)(hbase + k0 + 16);
#pragma unroll
                for (int r = 0; r < 4; ++r)
#pragma unroll
                    for (int s = 0; s < 2; ++s) {
                        const __bf16* p = uls[hf][s] + (size_t)r * (RNN_ * RNN_) + k0;
                        bb[r * 2 + s].h[0] = *(const bf16x8*)(p);
                        bb[r * 2 + s].h[1] = *(const bf16x8*)(p + 16);
                    }
#pragma unroll
                for (int r = 0; r < 4; ++r)
#pragma unroll
                    for (int s = 0; s < 2; ++s)
                        acc[r][s] = __builtin_amdgcn_wmma_f32_16x16x32_bf16(
                            false, a.v, false, bb[r * 2 + s].v, (short)0, acc[r][s], false, false);
            }

            // ---- gates + masked state update; write h_new into ping-pong buffer ----
#pragma unroll
            for (int s = 0; s < 2; ++s) {
                const int j = w * 64 + hf * 32 + s * 16 + lcol;
#pragma unroll
                for (int v = 0; v < 8; ++v) {
                    float ig = sigm(acc[0][s][v] + bia[hf][0][s]);
                    float fg = sigm(acc[1][s][v] + bia[hf][1][s]);
                    float gg = tanhf(acc[2][s][v] + bia[hf][2][s]);
                    float og = sigm(acc[3][s][v] + bia[hf][3][s]);
                    float cn = fg * c_[hf][s][v] + ig * gg;
                    float hn = og * tanhf(cn);
                    float m  = mk[v];
                    c_[hf][s][v] = m * cn + (1.0f - m) * c_[hf][s][v];
                    const int hidx = (v + mbase) * RNN_ + j;
                    float hp = bf2f(hsh[cur][hidx]);          // masked-keep previous h
                    float hv = m * hn + (1.0f - m) * hp;
                    p_[hf][s][v] += hv * pk[v];
                    hsh[nxt][hidx] = f2bf(hv);                // no race: nxt unread this step
                }
            }
        }
    }

    // write pooled states
#pragma unroll
    for (int hf = 0; hf < 2; ++hf)
#pragma unroll
        for (int s = 0; s < 2; ++s) {
            const int j = w * 64 + hf * 32 + s * 16 + lcol;
#pragma unroll
            for (int v = 0; v < 8; ++v)
                pooled[(size_t)(b0 + v + mbase) * RNN_ + j] = p_[hf][s][v];
        }
}

// ---------------- 4) MLP head ----------------
__global__ void mlp_relu_kernel(const float* __restrict__ in,   // (B, K)
                                const float* __restrict__ Wm,   // (K, N)
                                const float* __restrict__ bm,   // (N,)
                                float* __restrict__ out,        // (B, N)
                                int K, int N) {
    int idx = blockIdx.x * blockDim.x + threadIdx.x;            // over B*N
    int b = idx / N, n = idx - b * N;
    if (b >= B_) return;
    float acc = bm[n];
    const float* row = in + (size_t)b * K;
    for (int k = 0; k < K; ++k) acc = fmaf(row[k], Wm[(size_t)k * N + n], acc);
    out[idx] = fmaxf(acc, 0.0f);
}

__global__ void head_softmax_kernel(const float* __restrict__ h2,  // (B, HID)
                                    const float* __restrict__ Wc,  // (HID, LAB)
                                    const float* __restrict__ bc,  // (LAB,)
                                    float* __restrict__ out) {     // (B, LAB)
    int b = threadIdx.x;
    if (b >= B_) return;
    float lg[LAB_];
#pragma unroll
    for (int l = 0; l < LAB_; ++l) lg[l] = bc[l];
    const float* row = h2 + (size_t)b * HID_;
    for (int k = 0; k < HID_; ++k) {
        float hv = row[k];
#pragma unroll
        for (int l = 0; l < LAB_; ++l) lg[l] = fmaf(hv, Wc[k * LAB_ + l], lg[l]);
    }
    float mx = lg[0];
#pragma unroll
    for (int l = 1; l < LAB_; ++l) mx = fmaxf(mx, lg[l]);
    float sum = 0.0f;
#pragma unroll
    for (int l = 0; l < LAB_; ++l) { lg[l] = __expf(lg[l] - mx); sum += lg[l]; }
    float inv = 1.0f / sum;
#pragma unroll
    for (int l = 0; l < LAB_; ++l) out[b * LAB_ + l] = lg[l] * inv;
}

// ---------------- launcher ----------------
extern "C" void kernel_launch(void* const* d_in, const int* in_sizes, int n_in,
                              void* d_out, int out_size, void* d_ws, size_t ws_size,
                              hipStream_t stream) {
    const int*   x   = (const int*)  d_in[0];
    const float* emb = (const float*)d_in[1];
    const float* W   = (const float*)d_in[2];
    const float* U   = (const float*)d_in[3];
    const float* bg  = (const float*)d_in[4];
    const float* W1  = (const float*)d_in[5];
    const float* b1  = (const float*)d_in[6];
    const float* W2  = (const float*)d_in[7];
    const float* b2  = (const float*)d_in[8];
    const float* Wc  = (const float*)d_in[9];
    const float* bc  = (const float*)d_in[10];
    float* out = (float*)d_out;

    char* ws = (char*)d_ws;
    size_t off = 0;
    __bf16* eT = (__bf16*)(ws + off); off += (size_t)B_ * T_ * EMB_ * 2;      // 16 MB
    __bf16* Wt = (__bf16*)(ws + off); off += (size_t)NG_ * EMB_ * 2;          //  4 MB
    __bf16* Ut = (__bf16*)(ws + off); off += (size_t)NG_ * RNN_ * 2;          //  8 MB
    float* pooled = (float*)(ws + off); off += (size_t)B_ * RNN_ * 4;         // 256 KB
    float* h1 = (float*)(ws + off); off += (size_t)B_ * HID_ * 4;
    float* h2 = (float*)(ws + off); off += (size_t)B_ * HID_ * 4;

    // 1) embedding gather + bf16 cast  (B*T*EMB = 8388608 elems)
    gather_cast_kernel<<<(B_ * T_ * EMB_) / 256, 256, 0, stream>>>(x, emb, eT);
    // 2) weight transpose + cast
    transpose_cast_kernel<<<((size_t)EMB_ * NG_) / 256, 256, 0, stream>>>(W, Wt, EMB_, NG_);
    transpose_cast_kernel<<<((size_t)RNN_ * NG_) / 256, 256, 0, stream>>>(U, Ut, RNN_, NG_);
    // 3) recurrence (4 batch tiles of 16, persistent over all 256 steps)
    lstm_kernel<<<B_ / 16, 512, 0, stream>>>(x, eT, Wt, Ut, bg, pooled);
    // 4) MLP head + softmax
    mlp_relu_kernel<<<(B_ * HID_) / 256, 256, 0, stream>>>(pooled, W1, b1, h1, RNN_, HID_);
    mlp_relu_kernel<<<(B_ * HID_) / 256, 256, 0, stream>>>(h1, W2, b2, h2, HID_, HID_);
    head_softmax_kernel<<<1, 64, 0, stream>>>(h2, Wc, bc, out);
}
